// QNetwork_37160057045146
// MI455X (gfx1250) — matrix operand order repaired and checked by reading, not compiled, not run on hardware
//
#include <hip/hip_runtime.h>
#include <hip/hip_bf16.h>
#include <stdint.h>

typedef __attribute__((ext_vector_type(16))) _Float16 v16h;
typedef __attribute__((ext_vector_type(8)))  float    v8f;

#define WAVES_PER_BLOCK 8
#define KMAXF 256          // largest K used (conv5)
#define KPADF 4            // +4 floats/row -> 260-dword stride -> bank = 4*n, conflict-free

// ---------------------------------------------------------------------------
// WMMA GEMM: C[M,Ncols] = act(A[M,K] @ B + bias)
//  fp32 in/out, fp16 WMMA compute, fp32 accumulate.
//  Grid: x = M/(16*WAVES_PER_BLOCK) row-tiles, y = Ncols/16 col-strips.
//  B column strip (16 x K, f32) is staged into LDS with CDNA5 async
//  global->LDS copies (GLOBAL_LOAD_ASYNC_TO_LDS_B32, ASYNCcnt), then the
//  8 waves each produce one 16x16 tile with V_WMMA_F32_16X16X32_F16.
//  BTRANS=0: B is [K, Ncols] row-major; BTRANS=1: B is [Ncols, K] row-major.
// ---------------------------------------------------------------------------
template<bool BTRANS, bool USEBIAS, bool RELU>
__global__ void wmma_gemm_f16(const float* __restrict__ A,
                              const float* __restrict__ Braw,
                              const float* __restrict__ bias,
                              float* __restrict__ C,
                              int M, int Ncols, int K) {
    __shared__ float ldsB[16 * (KMAXF + KPADF)];

    const int tid  = threadIdx.x;
    const int lane = tid & 31;
    const int wave = tid >> 5;
    const int tn   = blockIdx.y;                  // column strip
    const int tm   = blockIdx.x * WAVES_PER_BLOCK + wave;

    // async-stage B strip: 16 cols x K rows of f32, direct global->LDS
    for (int idx = tid; idx < 16 * K; idx += blockDim.x) {
        const int n = idx >> 8;                   // k-major fill (K <= 256)
        const int k = idx & 255;
        if (k < K) {
            const float* src = BTRANS
                ? &Braw[(size_t)(tn * 16 + n) * K + k]
                : &Braw[(size_t)k * Ncols + (tn * 16 + n)];
            const unsigned ldsOff =
                (unsigned)(uintptr_t)(&ldsB[n * (KMAXF + KPADF) + k]);
            asm volatile("global_load_async_to_lds_b32 %0, %1, off"
                         :: "v"(ldsOff), "v"(src) : "memory");
        }
    }
    asm volatile("s_wait_asynccnt 0x0" ::: "memory");
    __syncthreads();

    if (tm >= (M >> 4)) return;                   // wave-uniform: EXEC stays all-1s

    const int g = lane >> 4;                      // lane group (0/1)
    const int m = lane & 15;                      // M index (A) / N index (B,C)

    v8f acc = {};
    const float* Arow = A + (size_t)(tm * 16 + m) * K;
    const float* Bln  = ldsB + m * (KMAXF + KPADF);

    for (int k0 = 0; k0 < K; k0 += 32) {
        __builtin_prefetch(Arow + k0 + 32, 0, 1); // speculative: next k-step
        v16h a, b;
        // A fragment: 16-bit A 16x32 layout (ISA 7.12.2)
#pragma unroll
        for (int h = 0; h < 16; ++h) {
            const int ka = h + ((h < 8) ? (g ? 8 : 0) : (g ? 16 : 8));
            a[h] = (_Float16)Arow[k0 + ka];
        }
        // B fragment from LDS: same K order as A (16-bit A 16x32 layout)
#pragma unroll
        for (int h = 0; h < 16; ++h) {
            b[h] = (_Float16)Bln[k0 + h + ((h < 8) ? (g << 3) : 8 + (g << 3))];  // ds_load_b128 x4 + cvt
        }
        acc = __builtin_amdgcn_wmma_f32_16x16x32_f16(
            false, a, false, b, (short)0, acc, false, false);
    }

    const int colc = tn * 16 + m;
    const float badd = USEBIAS ? bias[colc] : 0.0f;
#pragma unroll
    for (int r = 0; r < 8; ++r) {
        const int row = tm * 16 + r + (g << 3);   // C/D layout: M = r + 8*g
        float v = acc[r] + badd;
        if (RELU) v = fmaxf(v, 0.0f);
        C[(size_t)row * Ncols + colc] = v;
    }
}

// ---------------------------------------------------------------------------
// Elementwise helpers
// ---------------------------------------------------------------------------
__global__ void k_zero(float* __restrict__ p, long long n) {
    long long i = (long long)blockIdx.x * blockDim.x + threadIdx.x;
    if (i < n) p[i] = 0.0f;
}

__global__ void k_deg_count(const int* __restrict__ col, float* __restrict__ deg, int E) {
    int e = blockIdx.x * blockDim.x + threadIdx.x;
    if (e < E) atomicAdd(&deg[col[e]], 1.0f);
}

__global__ void k_dinv(const float* __restrict__ deg, float* __restrict__ dinv, int N) {
    int n = blockIdx.x * blockDim.x + threadIdx.x;
    if (n < N) dinv[n] = rsqrtf(deg[n] + 1.0f);
}

// agg[col] += dinv[row]*dinv[col] * xw[row]; one thread per (edge, 4 features)
__global__ void k_edge_agg4(const int* __restrict__ row, const int* __restrict__ col,
                            const float* __restrict__ dinv, const float* __restrict__ xw,
                            float* __restrict__ agg, int logF4, long long total4) {
    long long i = (long long)blockIdx.x * blockDim.x + threadIdx.x;
    if (i >= total4) return;
    const int F  = 4 << logF4;
    const int e  = (int)(i >> logF4);
    const int f4 = (int)(i & ((1 << logF4) - 1)) << 2;
    const int r = row[e], c = col[e];
    const float coef = dinv[r] * dinv[c];
    const float4 v = *(const float4*)&xw[(size_t)r * F + f4];
    float* dst = &agg[(size_t)c * F + f4];
    atomicAdd(dst + 0, coef * v.x);
    atomicAdd(dst + 1, coef * v.y);
    atomicAdd(dst + 2, coef * v.z);
    atomicAdd(dst + 3, coef * v.w);
}

// out = tanh(agg + dinv^2 * xw + bias)
__global__ void k_gcn_finish(const float* __restrict__ agg, const float* __restrict__ xw,
                             const float* __restrict__ dinv, const float* __restrict__ bias,
                             float* __restrict__ out, int logF, long long total) {
    long long i = (long long)blockIdx.x * blockDim.x + threadIdx.x;
    if (i >= total) return;
    const int F = 1 << logF;
    const int n = (int)(i >> logF);
    const int f = (int)(i & (F - 1));
    const float d = dinv[n];
    out[i] = tanhf(agg[i] + d * d * xw[i] + bias[f]);
}

// ---------------------------------------------------------------------------
// SortPool: per graph (128 nodes) rank by h[:,255] = x4[:,31] descending,
// select top-32, gather concat features into pooled[B,32,256]; emit indices.
// ---------------------------------------------------------------------------
__global__ void k_sortpool(const float* __restrict__ x1, const float* __restrict__ x2,
                           const float* __restrict__ x3, const float* __restrict__ x4,
                           float* __restrict__ pooled, float* __restrict__ topk_out) {
    __shared__ float key[128];
    __shared__ int sel[32];
    const int gph = blockIdx.x;
    const int t = threadIdx.x;
    const int node = gph * 128 + t;
    const float kv = x4[(size_t)node * 32 + 31];
    key[t] = kv;
    __syncthreads();
    int rank = 0;
    for (int j = 0; j < 128; ++j) {
        const float kj = key[j];
        rank += (kj > kv) || (kj == kv && j < t);   // stable argsort(-key)
    }
    if (rank < 32) {
        sel[rank] = t;
        topk_out[gph * 32 + rank] = (float)node;     // global index = idx + g*NPER
    }
    __syncthreads();
    for (int p = t; p < 32 * 256; p += 128) {
        const int r = p >> 8;
        const int f = p & 255;
        const int n = gph * 128 + sel[r];
        float v;
        if (f < 128)      v = x1[(size_t)n * 128 + f];
        else if (f < 192) v = x2[(size_t)n * 64 + (f - 128)];
        else if (f < 224) v = x3[(size_t)n * 32 + (f - 192)];
        else              v = x4[(size_t)n * 32 + (f - 224)];
        pooled[((size_t)gph * 32 + r) * 256 + f] = v;
    }
}

// ---------------------------------------------------------------------------
// Tail: maxpool(2) -> conv6 (K=5) + relu -> dense.  One workgroup per graph.
// z5 layout from conv5 GEMM: z5[(b*32 + l)*16 + oc] (already relu+bias).
// ---------------------------------------------------------------------------
__global__ void k_tail(const float* __restrict__ z5, const float* __restrict__ w6,
                       const float* __restrict__ b6, const float* __restrict__ dw,
                       const float* __restrict__ db, float* __restrict__ out) {
    __shared__ float mx[16][16];   // [pooled position][channel]
    __shared__ float y[384];
    const int gph = blockIdx.x;
    const int t = threadIdx.x;
    for (int p = t; p < 256; p += 128) {
        const int j = p >> 4, oc = p & 15;
        const float a = z5[((size_t)gph * 32 + 2 * j) * 16 + oc];
        const float b = z5[((size_t)gph * 32 + 2 * j + 1) * 16 + oc];
        mx[j][oc] = fmaxf(a, b);
    }
    __syncthreads();
    for (int p = t; p < 384; p += 128) {
        const int oc = p / 12;
        const int pos = p - oc * 12;
        float s = b6[oc];
#pragma unroll
        for (int ic = 0; ic < 16; ++ic)
#pragma unroll
            for (int tt = 0; tt < 5; ++tt)
                s += mx[pos + tt][ic] * w6[(oc * 16 + ic) * 5 + tt];
        y[oc * 12 + pos] = fmaxf(s, 0.0f);
    }
    __syncthreads();
    if (t < 27) {
        float s = db[t];
        for (int i = 0; i < 384; ++i) s += y[i] * dw[i * 27 + t];
        out[gph * 27 + t] = s;
    }
}

// ---------------------------------------------------------------------------
extern "C" void kernel_launch(void* const* d_in, const int* in_sizes, int n_in,
                              void* d_out, int out_size, void* d_ws, size_t ws_size,
                              hipStream_t stream) {
    const float* x     = (const float*)d_in[0];
    const int*   erow  = (const int*)d_in[1];
    const int    E     = in_sizes[1] / 2;
    const int*   ecol  = erow + E;
    const int    N     = in_sizes[0] / 128;     // 65536
    const int    B     = N / 128;               // 512

    const float* W[4]  = { (const float*)d_in[3], (const float*)d_in[5],
                           (const float*)d_in[7], (const float*)d_in[9] };
    const float* bb[4] = { (const float*)d_in[4], (const float*)d_in[6],
                           (const float*)d_in[8], (const float*)d_in[10] };
    const float* c5w = (const float*)d_in[11];
    const float* c5b = (const float*)d_in[12];
    const float* c6w = (const float*)d_in[13];
    const float* c6b = (const float*)d_in[14];
    const float* dW  = (const float*)d_in[15];
    const float* dB  = (const float*)d_in[16];

    // workspace layout (floats)
    float* ws    = (float*)d_ws;
    float* deg   = ws;                    // N
    float* dinv  = deg + N;               // N
    float* xw    = dinv + N;              // N*128 (max)
    float* agg   = xw + (size_t)N * 128;  // N*128 (max)
    float* xl[4];
    xl[0] = agg + (size_t)N * 128;        // N*128
    xl[1] = xl[0] + (size_t)N * 128;      // N*64
    xl[2] = xl[1] + (size_t)N * 64;       // N*32
    xl[3] = xl[2] + (size_t)N * 32;       // N*32
    float* pooled = xl[3] + (size_t)N * 32;        // B*32*256
    float* z5     = pooled + (size_t)B * 32 * 256; // (B*32)*16

    float* out_dense = (float*)d_out;               // [B,27]
    float* out_topk  = out_dense + (size_t)B * 27;  // [B,32] (stored as float)

    const int T = 256;
    // degrees -> dinv
    k_zero<<<(N + T - 1) / T, T, 0, stream>>>(deg, N);
    k_deg_count<<<(E + T - 1) / T, T, 0, stream>>>(ecol, deg, E);
    k_dinv<<<(N + T - 1) / T, T, 0, stream>>>(deg, dinv, N);

    // 4 GCN layers
    const int Fin[4]  = { 128, 128, 64, 32 };
    const int Fout[4] = { 128, 64, 32, 32 };
    const int logF[4] = { 7, 6, 5, 5 };
    const float* Aprev = x;
    for (int l = 0; l < 4; ++l) {
        const long long nf = (long long)N * Fout[l];
        // xw = A @ W   (WMMA, B strip async-staged in LDS)
        dim3 grid(N / (16 * WAVES_PER_BLOCK), Fout[l] / 16);
        wmma_gemm_f16<false, false, false>
            <<<grid, 32 * WAVES_PER_BLOCK, 0, stream>>>(Aprev, W[l], nullptr, xw,
                                                        N, Fout[l], Fin[l]);
        // agg = 0; scatter normalized neighbor features (float4 per thread)
        k_zero<<<(int)((nf + T - 1) / T), T, 0, stream>>>(agg, nf);
        const long long ef4 = (long long)E << (logF[l] - 2);
        k_edge_agg4<<<(int)((ef4 + T - 1) / T), T, 0, stream>>>(erow, ecol, dinv, xw, agg,
                                                                logF[l] - 2, ef4);
        // x_l = tanh(agg + dinv^2*xw + b)
        k_gcn_finish<<<(int)((nf + T - 1) / T), T, 0, stream>>>(agg, xw, dinv, bb[l], xl[l],
                                                               logF[l], nf);
        Aprev = xl[l];
    }

    // SortPool (writes pooled + top-k indices directly into d_out tail)
    k_sortpool<<<B, 128, 0, stream>>>(xl[0], xl[1], xl[2], xl[3], pooled, out_topk);

    // conv5 (stride==kernel==256) as GEMM: [B*32,256] @ c5w^T[256,16], +bias, relu
    {
        dim3 grid((B * 32) / (16 * WAVES_PER_BLOCK), 1);
        wmma_gemm_f16<true, true, true>
            <<<grid, 32 * WAVES_PER_BLOCK, 0, stream>>>(pooled, c5w, c5b, z5,
                                                        B * 32, 16, 256);
    }

    // maxpool + conv6 + dense
    k_tail<<<B, 128, 0, stream>>>(z5, c6w, c6b, dW, dB, out_dense);
}
